// VectorQuantizer_85770496901739
// MI455X (gfx1250) — matrix operand order repaired and checked by reading, not compiled, not run on hardware
//
#include <hip/hip_runtime.h>

typedef __attribute__((ext_vector_type(16))) __bf16 v16bf;
typedef __attribute__((ext_vector_type(8)))  float  v8f;
typedef unsigned int u32x4 __attribute__((ext_vector_type(4)));
typedef int          i32x4 __attribute__((ext_vector_type(4)));
typedef int          i32x8 __attribute__((ext_vector_type(8)));

#define B_   32
#define D_   256
#define T_   4096
#define K_   1024
#define BETA 0.25f
#define NROWS     (B_ * T_)        // 131072
#define TILE_ROWS 256              // rows per block (one row per thread)
#define CB_TILE   128              // codebook entries staged per LDS tile
#define ROW_PITCH 264              // bf16 elements (528 B) -> bank-conflict-free strides
#define NSTAGES   (K_ / CB_TILE)   // 8
#define NBLOCKS   (NROWS / TILE_ROWS) // 512

// LDS layout (bytes)
#define XS_OFF    0
#define CBS0_OFF  (TILE_ROWS * ROW_PITCH * 2)                 // 135168
#define CBS1_OFF  (CBS0_OFF + CB_TILE * ROW_PITCH * 2)        // 202752
#define EN_OFF    (CBS1_OFF + CB_TILE * ROW_PITCH * 2)        // 270336
#define XN_OFF    (EN_OFF + K_ * 4)                           // 274432
#define BIDX_OFF  (XN_OFF + TILE_ROWS * 4)                    // 275456
#define BDIST_OFF (BIDX_OFF + TILE_ROWS * 4)                  // 276480
#define LRED_OFF  (BDIST_OFF + TILE_ROWS * 4)                 // 277504
#define LDS_TOTAL (LRED_OFF + 256 * 4)                        // 278528

// Workspace layout (bytes)
#define WS_CBB    0                       // K_*D_ bf16 = 524288
#define WS_ENORM  (K_ * D_ * 2)           // 524288
#define WS_HIST   (WS_ENORM + K_ * 4)     // 528384
#define WS_LPART  (WS_HIST  + K_ * 4)     // 532480 (K_ slots; only NBLOCKS used, rest stay 0)

union Frag { uint4 q[2]; v16bf v; };

// ---------------------------------------------------------------------------
// Prep: codebook f32 -> bf16, per-entry ||e||^2, zero histogram / loss partials
// ---------------------------------------------------------------------------
__global__ __launch_bounds__(256)
void vq_prep(const float* __restrict__ cb, __bf16* __restrict__ cbb,
             float* __restrict__ enorm, unsigned* __restrict__ hist,
             float* __restrict__ lpart) {
    const int k = blockIdx.x;       // codebook entry
    const int d = threadIdx.x;      // dim (256 == D_)
    float c = cb[k * D_ + d];
    cbb[k * D_ + d] = (__bf16)c;
    float s = c * c;
    #pragma unroll
    for (int off = 16; off > 0; off >>= 1) s += __shfl_down(s, off, 32);
    __shared__ float wsum[8];
    const int lane = threadIdx.x & 31, w = threadIdx.x >> 5;
    if (lane == 0) wsum[w] = s;
    __syncthreads();
    if (threadIdx.x == 0) {
        float t = 0.f;
        #pragma unroll
        for (int i = 0; i < 8; ++i) t += wsum[i];
        enorm[k] = t;
        hist[k]  = 0u;
        lpart[k] = 0.f;
    }
}

// ---------------------------------------------------------------------------
// TDM: async-load one 128x512B codebook tile into LDS with 16B pad per 512B
// (row pitch 528 B == ROW_PITCH*2). D# per CDNA5 ISA ch.8 (data_size=4B).
// clang-23 toolchain: 6-arg builtin (g0 v4u32, g1 v8i32, v4i32, v4i32, v8i32, cpol)
// ---------------------------------------------------------------------------
__device__ __forceinline__
void tdm_load_cb_tile(const __bf16* gsrc, unsigned lds_byte_addr) {
    unsigned long long ga = (unsigned long long)(size_t)gsrc;
    u32x4 g0;
    g0[0] = 1u;                                        // count=1, user descriptor
    g0[1] = lds_byte_addr;                             // lds_addr
    g0[2] = (unsigned)(ga & 0xFFFFFFFFu);              // global_addr[31:0]
    g0[3] = (unsigned)((ga >> 32) & 0x01FFFFFFu)       // global_addr[56:32]
          | (2u << 30);                                // type=2 (image)
    i32x8 g1;
    g1[0] = (int)((2u << 16)                           // data_size: 4 bytes
          | (1u << 20)                                 // pad_enable
          | (6u << 22)                                 // pad_interval: 128 DWORDs
          | (3u << 25));                               // pad_amount: 4 DWORDs
    g1[1] = (int)(128u << 16);                         // tensor_dim0 = 128 DW (rows of 512B)
    g1[2] = (int)(128u << 16);                         // tensor_dim1 = 128 rows
    g1[3] = (int)(128u << 16);                         // tile_dim0 = 128 DW
    g1[4] = (int)(128u);                               // tile_dim1 = 128, tile_dim2 = 0
    g1[5] = (int)(128u);                               // tensor_dim0_stride = 128 DW
    g1[6] = 0;                                         // stride0 hi / stride1 lo
    g1[7] = 0;                                         // stride1 hi
    i32x4 gz4 = {0, 0, 0, 0};                          // groups 2/3: unused (2D tensor)
    i32x8 gz8 = {0, 0, 0, 0, 0, 0, 0, 0};
    __builtin_amdgcn_tensor_load_to_lds(g0, g1, gz4, gz4, gz8, 0);
}

// ---------------------------------------------------------------------------
// Main: fused distance GEMM (bf16 WMMA, 32 rows/wave) + argmin + gather + loss
// ---------------------------------------------------------------------------
__global__ __launch_bounds__(256)
void vq_main(const float* __restrict__ x, const float* __restrict__ cbf,
             const __bf16* __restrict__ cbb, const float* __restrict__ enormg,
             unsigned* __restrict__ hist, float* __restrict__ lpart,
             float* __restrict__ outq) {
    extern __shared__ char smem[];
    __bf16*  xs    = (__bf16*)(smem + XS_OFF);     // [256][264] = -2x (bf16)
    float*   en    = (float*)(smem + EN_OFF);      // [1024] ||e||^2
    float*   xnorm = (float*)(smem + XN_OFF);      // [256]  ||x||^2 (f32)
    int*     bidx  = (int*)(smem + BIDX_OFF);      // [256]
    float*   bdist = (float*)(smem + BDIST_OFF);   // [256]
    float*   lred  = (float*)(smem + LRED_OFF);    // [256]

    const int tid  = threadIdx.x;
    const int row0 = blockIdx.x * TILE_ROWS;
    const int b    = row0 / T_;
    const int t0   = row0 % T_;               // TILE_ROWS divides T_: tile stays in one b

    // ---- load x slab [D_, 256] strided-by-T, emit -2x bf16, accumulate ||x||^2
    float xs2 = 0.f;
    const float* xin = x + ((size_t)b * D_) * T_ + (size_t)t0 + tid;
    for (int dd = 0; dd < D_; ++dd) {
        float v = xin[(size_t)dd * T_];
        xs2 += v * v;
        xs[tid * ROW_PITCH + dd] = (__bf16)(-2.0f * v);
    }
    xnorm[tid] = xs2;
    for (int i = tid; i < K_; i += 256) en[i] = enormg[i];

    // ---- async TDM prologue: stage 0 -> cb buffer 0 (issued by wave 0 only)
    const int lane = tid & 31;
    const int w    = tid >> 5;                // wave id 0..7
    const unsigned cbs_lds[2] = {
        (unsigned)(size_t)(smem + CBS0_OFF),  // LDS aperture: addr[31:0] = LDS offset
        (unsigned)(size_t)(smem + CBS1_OFF)
    };
    if (w == 0) tdm_load_cb_tile(cbb, cbs_lds[0]);
    __syncthreads();

    // ---- build A fragments for this wave's 32 rows (two 16-row M-blocks)
    const int rowl0 = w * 32 + (lane & 15);   // M-block 0
    const char* xrow0 = (const char*)(xs + rowl0 * ROW_PITCH);
    const char* xrow1 = xrow0 + 16 * ROW_PITCH * 2;
    const int selA = (lane >> 4) * 16;        // byte select for K-half
    uint4 a0_lo[8], a0_hi[8], a1_lo[8], a1_hi[8];
    #pragma unroll
    for (int c = 0; c < 8; ++c) {
        a0_lo[c] = *(const uint4*)(xrow0 + c * 64 + selA);
        a0_hi[c] = *(const uint4*)(xrow0 + c * 64 + 32 + selA);
        a1_lo[c] = *(const uint4*)(xrow1 + c * 64 + selA);
        a1_hi[c] = *(const uint4*)(xrow1 + c * 64 + 32 + selA);
    }

    v8f  bestd0, bestd1;
    int  besti0[8], besti1[8];
    #pragma unroll
    for (int j = 0; j < 8; ++j) {
        bestd0[j] = 3.4e38f; besti0[j] = 0;
        bestd1[j] = 3.4e38f; besti1[j] = 0;
    }

    const int selB = (lane >> 4) * 16;        // byte select for B K-half

    for (int s = 0; s < NSTAGES; ++s) {
        __syncthreads();  // everyone done reading buf[(s+1)&1] (used in stage s-1)
        // kick next stage's TDM load into the other buffer, overlapped with compute
        if (w == 0 && s + 1 < NSTAGES)
            tdm_load_cb_tile(cbb + (size_t)(s + 1) * CB_TILE * D_, cbs_lds[(s + 1) & 1]);
        // preload this stage's ||e||^2 per column group (reg-resident seeds)
        float e2v[8];
        #pragma unroll
        for (int g = 0; g < 8; ++g)
            e2v[g] = en[s * CB_TILE + g * 16 + (lane & 15)];
        // wave 0: ensure stage-s tile has landed (TDM completes in order)
        if (w == 0) {
            if (s + 1 < NSTAGES) __builtin_amdgcn_s_wait_tensorcnt(1);
            else                 __builtin_amdgcn_s_wait_tensorcnt(0);
        }
        __syncthreads();

        const char* cbs = (const char*)smem + ((s & 1) ? CBS1_OFF : CBS0_OFF);

        // 8 column groups of 16 codebook entries each
        for (int g = 0; g < 8; ++g) {
            const int nglob = s * CB_TILE + g * 16 + (lane & 15);
            v8f acc0, acc1;
            #pragma unroll
            for (int j = 0; j < 8; ++j) { acc0[j] = e2v[g]; acc1[j] = e2v[g]; }
            const char* colp = cbs + (g * 16 + (lane & 15)) * (ROW_PITCH * 2) + selB;
            // software-pipelined B (double-buffered), each B reused by 2 WMMAs
            Frag bf[2];
            bf[0].q[0] = *(const uint4*)(colp);
            bf[0].q[1] = *(const uint4*)(colp + 32);
            #pragma unroll
            for (int c = 0; c < 8; ++c) {     // D=256 -> 8 WMMA k-steps
                if (c + 1 < 8) {
                    bf[(c + 1) & 1].q[0] = *(const uint4*)(colp + (c + 1) * 64);
                    bf[(c + 1) & 1].q[1] = *(const uint4*)(colp + (c + 1) * 64 + 32);
                }
                Frag af0, af1;
                af0.q[0] = a0_lo[c]; af0.q[1] = a0_hi[c];
                af1.q[0] = a1_lo[c]; af1.q[1] = a1_hi[c];
                acc0 = __builtin_amdgcn_wmma_f32_16x16x32_bf16(
                    false, af0.v, false, bf[c & 1].v, (short)0, acc0, false, false);
                acc1 = __builtin_amdgcn_wmma_f32_16x16x32_bf16(
                    false, af1.v, false, bf[c & 1].v, (short)0, acc1, false, false);
            }
            // acc[j] = ||e||^2 - 2 x.e
            #pragma unroll
            for (int j = 0; j < 8; ++j) {
                if (acc0[j] < bestd0[j]) { bestd0[j] = acc0[j]; besti0[j] = nglob; }
                if (acc1[j] < bestd1[j]) { bestd1[j] = acc1[j]; besti1[j] = nglob; }
            }
        }
    }

    // ---- argmin across the 16 lanes holding different entries (C layout: lane=N)
    #pragma unroll
    for (int m = 1; m < 16; m <<= 1) {
        #pragma unroll
        for (int j = 0; j < 8; ++j) {
            float od0 = __shfl_xor(bestd0[j], m, 16);
            int   oi0 = __shfl_xor(besti0[j], m, 16);
            if (od0 < bestd0[j] || (od0 == bestd0[j] && oi0 < besti0[j])) {
                bestd0[j] = od0; besti0[j] = oi0;
            }
            float od1 = __shfl_xor(bestd1[j], m, 16);
            int   oi1 = __shfl_xor(besti1[j], m, 16);
            if (od1 < bestd1[j] || (od1 == bestd1[j] && oi1 < besti1[j])) {
                bestd1[j] = od1; besti1[j] = oi1;
            }
        }
    }
    if ((lane & 15) == 0) {
        const int half = (lane >> 4) ? 8 : 0; // lanes 0-15: M=j, 16-31: M=8+j
        const int base0 = w * 32 + half;
        const int base1 = w * 32 + 16 + half;
        #pragma unroll
        for (int j = 0; j < 8; ++j) {
            bdist[base0 + j] = bestd0[j]; bidx[base0 + j] = besti0[j];
            bdist[base1 + j] = bestd1[j]; bidx[base1 + j] = besti1[j];
        }
    }
    __syncthreads();

    // ---- loss partial (||x||^2 + best_dist) and code-usage histogram
    lred[tid] = xnorm[tid] + bdist[tid];
    atomicAdd(&hist[bidx[tid]], 1u);
    __syncthreads();
    for (int s2 = 128; s2 > 0; s2 >>= 1) {
        if (tid < s2) lred[tid] += lred[tid + s2];
        __syncthreads();
    }
    if (tid == 0) lpart[blockIdx.x] = lred[0];

    // ---- quantized output: gather f32 codebook rows, write [B, D, T] coalesced in t
    const int myidx = bidx[tid];
    float* oq = outq + ((size_t)b * D_) * T_ + (size_t)t0 + tid;
    const float* crow = cbf + (size_t)myidx * D_;
    for (int dd = 0; dd < D_; ++dd) {
        oq[(size_t)dd * T_] = crow[dd];
    }
}

// ---------------------------------------------------------------------------
// Finalize: loss scalar and perplexity (deterministic fixed-order reduction)
// ---------------------------------------------------------------------------
__global__ __launch_bounds__(256)
void vq_final(const unsigned* __restrict__ hist, const float* __restrict__ lpart,
              float* __restrict__ out_loss, float* __restrict__ out_perp) {
    __shared__ float se[256], sl[256];
    float ent = 0.f, ls = 0.f;
    const float invn = 1.0f / (float)NROWS;
    for (int i = threadIdx.x; i < K_; i += 256) {
        float p = (float)hist[i] * invn;
        ent += p * __logf(p + 1e-10f);
        ls  += lpart[i];
    }
    se[threadIdx.x] = ent;
    sl[threadIdx.x] = ls;
    __syncthreads();
    for (int s = 128; s > 0; s >>= 1) {
        if (threadIdx.x < s) {
            se[threadIdx.x] += se[threadIdx.x + s];
            sl[threadIdx.x] += sl[threadIdx.x + s];
        }
        __syncthreads();
    }
    if (threadIdx.x == 0) {
        *out_perp = __expf(-se[0]);
        *out_loss = sl[0] * (1.0f + BETA) / (float)((size_t)NROWS * D_);
    }
}

// ---------------------------------------------------------------------------
extern "C" void kernel_launch(void* const* d_in, const int* in_sizes, int n_in,
                              void* d_out, int out_size, void* d_ws, size_t ws_size,
                              hipStream_t stream) {
    const float* x   = (const float*)d_in[0];   // [B, D, T] f32
    const float* cbf = (const float*)d_in[1];   // [K, D]    f32
    char* ws = (char*)d_ws;
    __bf16*   cbb   = (__bf16*)(ws + WS_CBB);
    float*    enorm = (float*)(ws + WS_ENORM);
    unsigned* hist  = (unsigned*)(ws + WS_HIST);
    float*    lpart = (float*)(ws + WS_LPART);
    float*    out   = (float*)d_out;            // [loss, quantized(B*D*T), perplexity]

    vq_prep<<<K_, 256, 0, stream>>>(cbf, cbb, enorm, hist, lpart);
    vq_main<<<NBLOCKS, 256, LDS_TOTAL, stream>>>(x, cbf, cbb, enorm, hist, lpart, out + 1);
    vq_final<<<1, 256, 0, stream>>>(hist, lpart, out, out + 1 + (size_t)B_ * D_ * T_);
}
